// ASTScannedGRU_7773890806119
// MI455X (gfx1250) — matrix operand order repaired and checked by reading, not compiled, not run on hardware
//
#include <hip/hip_runtime.h>
#include <hip/hip_bf16.h>

// ---------------------------------------------------------------------------
// Scanned GRU for MI455X (gfx1250, wave32, WMMA).
//   T=512, B=128, D=512, H=512, gates 3H=1536
// One persistent kernel runs the whole scan; per time-step it fuses
//   xg_t = x_t @ Wi + bi   and   hg_t = mask(h) @ Wh
// as bf16 WMMA (v_wmma_f32_16x16x32_bf16) with f32 accumulators, then the
// gate math. Grid-wide sync via global atomics (graph-capture safe).
// Weights pre-transposed to bf16 [3H][K] in d_ws so each lane's B fragment is
// two contiguous global_load_b128; A tiles staged in LDS per the ISA 16-bit
// A-matrix layout. Inner GEMM loop is software-pipelined (double-buffered
// fragments) so WMMAs overlap the next iteration's loads.
// ---------------------------------------------------------------------------

typedef __bf16 bf16_t;
typedef __attribute__((ext_vector_type(4)))  __bf16 v4bf;
typedef __attribute__((ext_vector_type(8)))  __bf16 v8bf;
typedef __attribute__((ext_vector_type(16))) __bf16 v16bf;
typedef __attribute__((ext_vector_type(8)))  float  v8f;

#define T_STEPS 512
#define BB      128
#define DD      512
#define HH      512
#define G3      1536
#define NBLK    96      // 8 M-tiles x 12 N-groups; must be co-resident
#define NTHR    256     // 8 waves of 32

// ---------------- grid-wide barrier (persistent-kernel sync) ----------------
__device__ __forceinline__ void grid_sync(unsigned* bar) {
    __threadfence();
    __syncthreads();
    if (threadIdx.x == 0) {
        unsigned gen = __hip_atomic_load(bar + 1, __ATOMIC_RELAXED, __HIP_MEMORY_SCOPE_AGENT);
        unsigned arrived = __hip_atomic_fetch_add(bar, 1u, __ATOMIC_ACQ_REL, __HIP_MEMORY_SCOPE_AGENT);
        if (arrived == NBLK - 1u) {
            __hip_atomic_store(bar, 0u, __ATOMIC_RELAXED, __HIP_MEMORY_SCOPE_AGENT);
            __hip_atomic_fetch_add(bar + 1, 1u, __ATOMIC_RELEASE, __HIP_MEMORY_SCOPE_AGENT);
        } else {
            while (__hip_atomic_load(bar + 1, __ATOMIC_ACQUIRE, __HIP_MEMORY_SCOPE_AGENT) == gen) {
                __builtin_amdgcn_s_sleep(1);
            }
        }
    }
    __syncthreads();
}

__device__ __forceinline__ v16bf mk16(v8bf lo, v8bf hi) {
    return __builtin_shufflevector(lo, hi,
        0,1,2,3,4,5,6,7,8,9,10,11,12,13,14,15);
}

// ---------------- init kernels ----------------
// W [K][N] f32 row-major  ->  WT [N][K] bf16 row-major (K-contiguous per column)
__global__ void k_convert_w(const float* __restrict__ W, bf16_t* __restrict__ WT,
                            int K, int N) {
    int i = blockIdx.x * blockDim.x + threadIdx.x;      // i < K*N
    int k = i / N;
    int n = i - k * N;
    WT[(size_t)n * K + k] = (bf16_t)W[i];
}

__global__ void k_init(const float* __restrict__ h0, float* __restrict__ h,
                       unsigned* __restrict__ bar) {
    int i = blockIdx.x * blockDim.x + threadIdx.x;      // 65536 threads = B*H
    h[i] = h0[i];
    if (i == 0) { bar[0] = 0u; bar[1] = 0u; }
}

// ---------------- persistent scan kernel ----------------
__global__ __launch_bounds__(NTHR) void gru_scan(
    const float* __restrict__ x,      // [T,B,D]
    const int*   __restrict__ resets, // [T,B]
    const bf16_t* __restrict__ WiT,   // [3H][D] bf16
    const bf16_t* __restrict__ WhT,   // [3H][H] bf16
    const float* __restrict__ bi,     // [3H]
    const float* __restrict__ bhn,    // [H]
    float* __restrict__ h,            // [B,H]   carry (ws)
    float* __restrict__ xg,           // [B,3H]  scratch (ws)
    float* __restrict__ hg,           // [B,3H]  scratch (ws)
    unsigned* __restrict__ bar,
    float* __restrict__ out)          // [B*H h_final | T*B*H ys]
{
    __shared__ bf16_t ldsX[16 * DD];  // 16KB: rows of x_t (bf16)
    __shared__ bf16_t ldsH[16 * HH];  // 16KB: rows of masked h (bf16)

    const int tid  = threadIdx.x;
    const int lane = tid & 31;
    const int wave = tid >> 5;                // 0..7
    const int mt   = blockIdx.x & 7;          // M tile (B/16)
    const int ng   = blockIdx.x >> 3;         // 0..11
    const int m0   = mt * 16;
    const int n0   = ng * 128 + wave * 16;    // 16-wide column strip of 3H
    const int half = lane >> 4;               // lane group 0/1
    const int nn   = lane & 15;

    const int nthreads = NBLK * NTHR;
    const int gid      = blockIdx.x * NTHR + tid;

    // per-lane weight row bases (K-contiguous bf16), B-layout: chunks at k0+8*half, k0+16+8*half
    const bf16_t* bxRow = WiT + (size_t)(n0 + nn) * DD + half * 8;
    const bf16_t* bhRow = WhT + (size_t)(n0 + nn) * HH + half * 8;
    // per-lane A row base in LDS, ISA A-layout: chunks at k0+8*half, k0+16+8*half
    const int aOff = (lane & 15) * DD + half * 8;

    for (int t = 0; t < T_STEPS; ++t) {
        // ---- stage A tiles into LDS as bf16 (x_t rows + reset-masked h rows)
        const float* xbase = x + ((size_t)t * BB + m0) * DD;
        #pragma unroll
        for (int i = 0; i < 8; ++i) {
            int idx = i * NTHR + tid;          // 0..2047
            int r   = idx >> 7;                // 0..15
            int c   = (idx & 127) * 4;         // 0..508
            float4 vx = *(const float4*)(xbase + (size_t)r * DD + c);
            v4bf px = { (bf16_t)vx.x, (bf16_t)vx.y, (bf16_t)vx.z, (bf16_t)vx.w };
            *(v4bf*)(ldsX + r * DD + c) = px;

            float4 vh = *(const float4*)(h + (size_t)(m0 + r) * HH + c);
            if (resets[(size_t)t * BB + m0 + r]) vh = make_float4(0.f, 0.f, 0.f, 0.f);
            v4bf ph = { (bf16_t)vh.x, (bf16_t)vh.y, (bf16_t)vh.z, (bf16_t)vh.w };
            *(v4bf*)(ldsH + r * HH + c) = ph;
        }
        __syncthreads();

        // ---- software-pipelined WMMA: K = 512, 16 k-steps, double-buffered
        v8f accx = {};
        v8f acch = {};

        v16bf ax0 = mk16(*(const v8bf*)(ldsX + aOff),
                         *(const v8bf*)(ldsX + aOff + 16));
        v16bf bx0 = mk16(*(const v8bf*)(bxRow),
                         *(const v8bf*)(bxRow + 16));
        v16bf ah0 = mk16(*(const v8bf*)(ldsH + aOff),
                         *(const v8bf*)(ldsH + aOff + 16));
        v16bf bh0 = mk16(*(const v8bf*)(bhRow),
                         *(const v8bf*)(bhRow + 16));

        #pragma unroll 3
        for (int k0 = 32; k0 < DD; k0 += 32) {
            // issue next iteration's loads before consuming current fragments
            v16bf ax1 = mk16(*(const v8bf*)(ldsX + aOff + k0),
                             *(const v8bf*)(ldsX + aOff + k0 + 16));
            v16bf bx1 = mk16(*(const v8bf*)(bxRow + k0),
                             *(const v8bf*)(bxRow + k0 + 16));
            v16bf ah1 = mk16(*(const v8bf*)(ldsH + aOff + k0),
                             *(const v8bf*)(ldsH + aOff + k0 + 16));
            v16bf bh1 = mk16(*(const v8bf*)(bhRow + k0),
                             *(const v8bf*)(bhRow + k0 + 16));
            __builtin_prefetch(bxRow + k0 + 192, 0, 3);   // global_prefetch_b8 near
            __builtin_prefetch(bhRow + k0 + 192, 0, 3);

            accx = __builtin_amdgcn_wmma_f32_16x16x32_bf16(
                false, ax0, false, bx0, (short)0, accx, false, false);
            acch = __builtin_amdgcn_wmma_f32_16x16x32_bf16(
                false, ah0, false, bh0, (short)0, acch, false, false);

            ax0 = ax1; bx0 = bx1; ah0 = ah1; bh0 = bh1;
        }
        accx = __builtin_amdgcn_wmma_f32_16x16x32_bf16(
            false, ax0, false, bx0, (short)0, accx, false, false);
        acch = __builtin_amdgcn_wmma_f32_16x16x32_bf16(
            false, ah0, false, bh0, (short)0, acch, false, false);

        // ---- write 16x16 D tiles (ISA layout: M = r + 8*half, N = lane&15)
        const int col = n0 + nn;
        const float bic = bi[col];
        #pragma unroll
        for (int r = 0; r < 8; ++r) {
            int m = m0 + r + half * 8;
            xg[(size_t)m * G3 + col] = accx[r] + bic;
            hg[(size_t)m * G3 + col] = acch[r];
        }

        grid_sync(bar);

        // ---- elementwise gates, update carry, emit ys[t]
        for (int i = gid; i < BB * HH; i += nthreads) {
            int b = i >> 9;             // /H
            int j = i & (HH - 1);
            size_t g = (size_t)b * G3 + j;
            float r_ = xg[g]          + hg[g];
            float z_ = xg[g + HH]     + hg[g + HH];
            float n_ = xg[g + 2 * HH];
            float hn = hg[g + 2 * HH];
            r_ = 1.f / (1.f + __expf(-r_));
            z_ = 1.f / (1.f + __expf(-z_));
            n_ = tanhf(n_ + r_ * (hn + bhn[j]));
            float hold = resets[(size_t)t * BB + b] ? 0.f : h[i];
            float hnew = (1.f - z_) * n_ + z_ * hold;
            h[i] = hnew;
            out[(size_t)(BB * HH) + (size_t)t * BB * HH + i] = hnew;  // ys[t]
            if (t == T_STEPS - 1) out[i] = hnew;                      // h_final
        }

        grid_sync(bar);
    }
}

// ---------------------------------------------------------------------------
extern "C" void kernel_launch(void* const* d_in, const int* in_sizes, int n_in,
                              void* d_out, int out_size, void* d_ws, size_t ws_size,
                              hipStream_t stream) {
    const float* x   = (const float*)d_in[0];   // [T,B,D]
    const int*   rst = (const int*)  d_in[1];   // [T,B]
    const float* h0  = (const float*)d_in[2];   // [B,H]
    const float* Wi  = (const float*)d_in[3];   // [D,3H]
    const float* bi  = (const float*)d_in[4];   // [3H]
    const float* Wh  = (const float*)d_in[5];   // [H,3H]
    const float* bhn = (const float*)d_in[6];   // [H]
    float* out = (float*)d_out;

    // workspace layout (bytes)
    char* ws = (char*)d_ws;
    bf16_t*   WiT = (bf16_t*)(ws);                         // 3H*D*2  = 1.5 MB
    bf16_t*   WhT = (bf16_t*)(ws + 1572864);               // 3H*H*2  = 1.5 MB
    float*    h   = (float*) (ws + 3145728);               // B*H*4   = 256 KB
    float*    xg  = (float*) (ws + 3407872);               // B*3H*4  = 768 KB
    float*    hg  = (float*) (ws + 4194304);               // B*3H*4  = 768 KB
    unsigned* bar = (unsigned*)(ws + 4980736);             // 8 B

    // bf16 transpose of both weight matrices: [K][3H] f32 -> [3H][K] bf16
    k_convert_w<<<(DD * G3) / 256, 256, 0, stream>>>(Wi, WiT, DD, G3);
    k_convert_w<<<(HH * G3) / 256, 256, 0, stream>>>(Wh, WhT, HH, G3);
    // carry init + barrier reset (runs every call -> deterministic replays)
    k_init<<<(BB * HH) / 256, 256, 0, stream>>>(h0, h, bar);

    gru_scan<<<NBLK, NTHR, 0, stream>>>(x, rst, WiT, WhT, bi, bhn,
                                        h, xg, hg, bar, out);
}
